// LinearRNN_19456201851451
// MI455X (gfx1250) — compile-verified, matrix operand order checked by reading, not run on hardware
//
#include <hip/hip_runtime.h>
#include <stdint.h>

// ---------------------------------------------------------------------------
// LinearRNN on MI455X (gfx1250). bf16 hi/lo split (3x v_wmma_f32_16x16x32_bf16
// per fp32 MAC tile, fp32 accumulate). v4: gemm_xw staging is software-
// pipelined -- x and B-slice global loads for kb+1 are issued into registers
// before the barrier, overlapping global latency with the 24-WMMA compute
// phase of kb. Conversion stays outside the K loops; fragments live in LDS.
// ---------------------------------------------------------------------------

typedef __attribute__((ext_vector_type(16))) __bf16    bf16x16;
typedef __attribute__((ext_vector_type(8)))  float     f32x8;
typedef __attribute__((ext_vector_type(4)))  uint32_t  u32x4;
typedef __attribute__((ext_vector_type(2)))  float     f32x2;

union BFrag {
  bf16x16  v;
  uint32_t u[8];
  u32x4    q[2];
};

__device__ __forceinline__ uint32_t f2bf_bits(float f) {
  uint32_t x = __float_as_uint(f);
  return (x + 0x7FFFu + ((x >> 16) & 1u)) >> 16;   // round-to-nearest-even
}
__device__ __forceinline__ float bf2f(uint32_t b) {
  return __uint_as_float(b << 16);
}
__device__ __forceinline__ void split_pack(float f0, float f1,
                                           uint32_t& hi, uint32_t& lo) {
  uint32_t h0 = f2bf_bits(f0), h1 = f2bf_bits(f1);
  float l0 = f0 - bf2f(h0);
  float l1 = f1 - bf2f(h1);
  hi = h0 | (h1 << 16);
  lo = f2bf_bits(l0) | (f2bf_bits(l1) << 16);
}

// Fragment order (ISA 7.12.2, 16-bit A/B): half 0 -> K {0..7,16..23},
// half 1 -> K {8..15,24..31}; within a half the 16 values are contiguous.
__device__ __forceinline__ int fr_half(int kk) { return (kk >> 3) & 1; }
__device__ __forceinline__ int fr_j(int kk)    { return (kk & 7) + (kk >= 16 ? 8 : 0); }
// inverse: kk = half*8 + j + (j>=8 ? 8 : 0)

// A-fragment LDS layout for a 16-row tile: [kb][row][half][16] ushorts.
__device__ __forceinline__ int aidx16(int kb, int row, int half, int j) {
  return (((kb * 16 + row) * 2 + half) << 4) + j;
}

__device__ __forceinline__ void store_frag1(unsigned short* fhi,
                                            unsigned short* flo,
                                            int row, int k, float f) {
  int kb = k >> 5, kk = k & 31;
  int idx = aidx16(kb, row, fr_half(kk), fr_j(kk));
  uint32_t h = f2bf_bits(f);
  float l = f - bf2f(h);
  fhi[idx] = (unsigned short)h;
  flo[idx] = (unsigned short)f2bf_bits(l);
}

__device__ __forceinline__ void load_Afrag(const unsigned short* fhi,
                                           const unsigned short* flo,
                                           int kb, int lane,
                                           BFrag& ahi, BFrag& alo) {
  int idx = aidx16(kb, lane & 15, (lane >> 4) & 1, 0);
  const u32x4* ph = (const u32x4*)(fhi + idx);
  const u32x4* pl = (const u32x4*)(flo + idx);
  ahi.q[0] = ph[0]; ahi.q[1] = ph[1];
  alo.q[0] = pl[0]; alo.q[1] = pl[1];
}

// B fragments pre-packed in global: [kb][col][half][16] ushorts.
__device__ __forceinline__ void load_B(const unsigned short* __restrict__ hi,
                                       const unsigned short* __restrict__ lo,
                                       int kb, int col, int half,
                                       BFrag& bhi, BFrag& blo) {
  size_t off = ((size_t)(kb * 512 + col) << 5) + (half << 4);
  const u32x4* ph = (const u32x4*)(hi + off);
  const u32x4* pl = (const u32x4*)(lo + off);
  bhi.q[0] = ph[0]; bhi.q[1] = ph[1];
  blo.q[0] = pl[0]; blo.q[1] = pl[1];
}

__device__ __forceinline__ f32x8 mm3(const BFrag& ahi, const BFrag& alo,
                                     const BFrag& bhi, const BFrag& blo,
                                     f32x8 acc) {
  acc = __builtin_amdgcn_wmma_f32_16x16x32_bf16(false, ahi.v, false, bhi.v,
                                                (short)0, acc, false, false);
  acc = __builtin_amdgcn_wmma_f32_16x16x32_bf16(false, ahi.v, false, blo.v,
                                                (short)0, acc, false, false);
  acc = __builtin_amdgcn_wmma_f32_16x16x32_bf16(false, alo.v, false, bhi.v,
                                                (short)0, acc, false, false);
  return acc;
}

// ---------------------------------------------------------------------------
// Kernel 1: split + fragment-order weights.  B[k][n] = W[n][k]  (x @ W^T).
// ---------------------------------------------------------------------------
__global__ __launch_bounds__(256) void prep_kernel(
    const float* __restrict__ Wxh, const float* __restrict__ Whh,
    unsigned short* __restrict__ xh_hi, unsigned short* __restrict__ xh_lo,
    unsigned short* __restrict__ hh_hi, unsigned short* __restrict__ hh_lo) {
  int gid = blockIdx.x * 256 + threadIdx.x;      // 0 .. 512*512-1
  int k = gid & 511, n = gid >> 9;
  int kb = k >> 5, kk = k & 31;
  size_t fidx = ((size_t)(kb * 512 + n) << 5) + (fr_half(kk) << 4) + fr_j(kk);

  float w1 = Wxh[(size_t)n * 512 + k];
  float w2 = Whh[(size_t)n * 512 + k];
  uint32_t h1 = f2bf_bits(w1), h2 = f2bf_bits(w2);
  xh_hi[fidx] = (unsigned short)h1;
  xh_lo[fidx] = (unsigned short)f2bf_bits(w1 - bf2f(h1));
  hh_hi[fidx] = (unsigned short)h2;
  hh_lo[fidx] = (unsigned short)f2bf_bits(w2 - bf2f(h2));
}

// ---------------------------------------------------------------------------
// Kernel 2: xW = x @ Wxh^T.  M=65536, N=512, K=512.
// WG tile 128(M) x 128(N); 8 waves as (mi in 0..3) x (nj in 0..1); each wave
// computes 32x64 of C (24 WMMA per kb).  Staging is register-prefetched one
// kb ahead so global latency overlaps the WMMA phase.
// ---------------------------------------------------------------------------
__global__ __launch_bounds__(256) void gemm_xw_kernel(
    const float* __restrict__ x,
    const unsigned short* __restrict__ wf_hi,
    const unsigned short* __restrict__ wf_lo,
    float* __restrict__ out) {
  __shared__ unsigned short GAhi[128 * 2 * 16];   // [row][half][16]  8KB
  __shared__ unsigned short GAlo[128 * 2 * 16];
  __shared__ unsigned short GBhi[128 * 2 * 16];   // [cl][half][16]   8KB
  __shared__ unsigned short GBlo[128 * 2 * 16];
  const int tid = threadIdx.x, lane = tid & 31, wv = tid >> 5;
  const int mi = wv & 3, nj = wv >> 2;
  const int m0 = blockIdx.x * 128;
  const int nbase = blockIdx.y * 128;             // WG column base
  const int n0 = nbase + nj * 64;                 // wave column base

  f32x8 acc[2][4] = {};
  f32x2 xr[8];                                    // prefetched x pairs
  u32x4 bhr[2], blr[2];                           // prefetched B slice

  // prologue: fetch kb = 0
  {
#pragma unroll
    for (int i = 0; i < 8; ++i) {
      int pi = tid + i * 256;
      int r = pi >> 4, kpair = (pi & 15) << 1;
      xr[i] = *(const f32x2*)(x + (size_t)(m0 + r) * 512 + kpair);
    }
    const u32x4* gh = (const u32x4*)(wf_hi + ((size_t)nbase << 5));
    const u32x4* gl = (const u32x4*)(wf_lo + ((size_t)nbase << 5));
    bhr[0] = gh[tid]; bhr[1] = gh[tid + 256];
    blr[0] = gl[tid]; blr[1] = gl[tid + 256];
  }

  for (int kb = 0; kb < 16; ++kb) {
    __syncthreads();                              // LDS free of readers
    // ---- store stage kb (from prefetched registers)
#pragma unroll
    for (int i = 0; i < 8; ++i) {
      int pi = tid + i * 256;
      int r = pi >> 4, kpair = (pi & 15) << 1;
      int idx = ((r * 2 + fr_half(kpair)) << 4) + fr_j(kpair);
      uint32_t hi, lo;
      split_pack(xr[i].x, xr[i].y, hi, lo);
      *(uint32_t*)(GAhi + idx) = hi;
      *(uint32_t*)(GAlo + idx) = lo;
    }
    {
      u32x4* dh = (u32x4*)GBhi;
      u32x4* dl = (u32x4*)GBlo;
      dh[tid]       = bhr[0];
      dh[tid + 256] = bhr[1];
      dl[tid]       = blr[0];
      dl[tid + 256] = blr[1];
    }
    // ---- prefetch kb+1: loads fly across the barrier and the WMMA phase
    if (kb < 15) {
      int kn = kb + 1;
#pragma unroll
      for (int i = 0; i < 8; ++i) {
        int pi = tid + i * 256;
        int r = pi >> 4, kpair = (pi & 15) << 1;
        xr[i] = *(const f32x2*)(x + (size_t)(m0 + r) * 512 + kn * 32 + kpair);
      }
      const u32x4* gh = (const u32x4*)(wf_hi + ((size_t)(kn * 512 + nbase) << 5));
      const u32x4* gl = (const u32x4*)(wf_lo + ((size_t)(kn * 512 + nbase) << 5));
      bhr[0] = gh[tid]; bhr[1] = gh[tid + 256];
      blr[0] = gl[tid]; blr[1] = gl[tid + 256];
    }
    __syncthreads();

    // ---- compute: 2 A row-blocks x 4 B tiles (24 WMMA)
    BFrag a0h, a0l, a1h, a1l;
    {
      int idx0 = ((mi * 32 + (lane & 15)) * 2 + ((lane >> 4) & 1)) << 4;
      int idx1 = idx0 + (16 * 2 * 16);            // +16 rows
      const u32x4* p;
      p = (const u32x4*)(GAhi + idx0); a0h.q[0] = p[0]; a0h.q[1] = p[1];
      p = (const u32x4*)(GAlo + idx0); a0l.q[0] = p[0]; a0l.q[1] = p[1];
      p = (const u32x4*)(GAhi + idx1); a1h.q[0] = p[0]; a1h.q[1] = p[1];
      p = (const u32x4*)(GAlo + idx1); a1l.q[0] = p[0]; a1l.q[1] = p[1];
    }
#pragma unroll
    for (int t = 0; t < 4; ++t) {
      int cl = nj * 64 + t * 16 + (lane & 15);    // column within WG slice
      int idx = ((cl * 2 + (lane >> 4)) << 4);
      BFrag bhi, blo;
      const u32x4* p;
      p = (const u32x4*)(GBhi + idx); bhi.q[0] = p[0]; bhi.q[1] = p[1];
      p = (const u32x4*)(GBlo + idx); blo.q[0] = p[0]; blo.q[1] = p[1];
      acc[0][t] = mm3(a0h, a0l, bhi, blo, acc[0][t]);
      acc[1][t] = mm3(a1h, a1l, bhi, blo, acc[1][t]);
    }
  }

  // C layout: lanes 0-15 -> N=lane, M=j; lanes 16-31 -> N=lane-16, M=8+j.
#pragma unroll
  for (int s = 0; s < 2; ++s) {
    const int mbase = m0 + mi * 32 + s * 16 + (lane >> 4) * 8;
#pragma unroll
    for (int t = 0; t < 4; ++t) {
      int col = n0 + t * 16 + (lane & 15);
#pragma unroll
      for (int j = 0; j < 8; ++j)
        out[(size_t)(mbase + j) * 512 + col] = acc[s][t][j];
    }
  }
}

// ---------------------------------------------------------------------------
// Kernel 3: recurrence, in place over rec (holds xW).  16 WGs x 16 rows.
// h lives in LDS as bf16 hi/lo fragments only (32KB), converted once/step.
// ---------------------------------------------------------------------------
__global__ __launch_bounds__(256) void rec_kernel(
    const unsigned short* __restrict__ wf_hi,
    const unsigned short* __restrict__ wf_lo,
    float* __restrict__ rec) {
  __shared__ unsigned short Fhi[16 * 16 * 2 * 16];   // [kb][row][half][16]
  __shared__ unsigned short Flo[16 * 16 * 2 * 16];
  const int tid = threadIdx.x, lane = tid & 31, wv = tid >> 5;
  const int r0 = blockIdx.x * 16;
  const int n0 = wv * 64;

  for (int i = tid; i < 16 * 16 * 2 * 16; i += 256) { Fhi[i] = 0; Flo[i] = 0; }
  __syncthreads();

  for (int t = 0; t < 256; ++t) {
    f32x8 acc[4] = {};
    for (int kb = 0; kb < 16; ++kb) {
      BFrag ahi, alo;
      load_Afrag(Fhi, Flo, kb, lane, ahi, alo);
#pragma unroll
      for (int tt = 0; tt < 4; ++tt) {
        BFrag bhi, blo;
        load_B(wf_hi, wf_lo, kb, n0 + tt * 16 + (lane & 15), lane >> 4,
               bhi, blo);
        acc[tt] = mm3(ahi, alo, bhi, blo, acc[tt]);
      }
    }
    __syncthreads();                              // fragment reads complete
    const int mloc = (lane >> 4) * 8;
#pragma unroll
    for (int tt = 0; tt < 4; ++tt) {
      int col = n0 + tt * 16 + (lane & 15);
#pragma unroll
      for (int j = 0; j < 8; ++j) {
        size_t go = ((size_t)(r0 + mloc + j) * 256 + t) * 512 + col;
        float v = acc[tt][j] + rec[go];           // + xW_t (in place)
        rec[go] = v;                              // h_t -> output
        store_frag1(Fhi, Flo, mloc + j, col, v);  // h_t -> next step (bf16)
      }
    }
    __syncthreads();
  }
}

// ---------------------------------------------------------------------------
// Kernel 4: 96 prediction steps + LayerNorm.  h as fragments only; LN stats
// reconstruct hi+lo from the fragment LDS (error ~2^-16, negligible for LN).
// ---------------------------------------------------------------------------
__global__ __launch_bounds__(256) void pred_kernel(
    const unsigned short* __restrict__ wf_hi,
    const unsigned short* __restrict__ wf_lo,
    const float* __restrict__ gamma, const float* __restrict__ beta,
    const float* __restrict__ rec, float* __restrict__ outs) {
  __shared__ unsigned short Fhi[16 * 16 * 2 * 16];
  __shared__ unsigned short Flo[16 * 16 * 2 * 16];
  __shared__ float red_s[16][17];
  __shared__ float red_q[16][17];
  __shared__ float mu_s[16], rs_s[16];
  const int tid = threadIdx.x, lane = tid & 31, wv = tid >> 5;
  const int r0 = blockIdx.x * 16;
  const int n0 = wv * 64;
  const int lrow = tid & 15, seg = tid >> 4;      // seg == kb for LN

  for (int li = tid; li < 16 * 512; li += 256) {  // h = rec[:, t=255, :]
    int r = li >> 9, c = li & 511;
    store_frag1(Fhi, Flo, r, c,
                rec[((size_t)(r0 + r) * 256 + 255) * 512 + c]);
  }
  __syncthreads();

  for (int t = 0; t < 96; ++t) {
    f32x8 acc[4] = {};
    for (int kb = 0; kb < 16; ++kb) {
      BFrag ahi, alo;
      load_Afrag(Fhi, Flo, kb, lane, ahi, alo);
#pragma unroll
      for (int tt = 0; tt < 4; ++tt) {
        BFrag bhi, blo;
        load_B(wf_hi, wf_lo, kb, n0 + tt * 16 + (lane & 15), lane >> 4,
               bhi, blo);
        acc[tt] = mm3(ahi, alo, bhi, blo, acc[tt]);
      }
    }
    __syncthreads();
    const int mloc = (lane >> 4) * 8;
#pragma unroll
    for (int tt = 0; tt < 4; ++tt) {
      int col = n0 + tt * 16 + (lane & 15);
#pragma unroll
      for (int j = 0; j < 8; ++j)
        store_frag1(Fhi, Flo, mloc + j, col, acc[tt][j]);  // h <- h@Whh^T
    }
    __syncthreads();

    // LayerNorm partials: thread (lrow, seg) scans kb=seg of row lrow.
    float s = 0.f, q = 0.f;
#pragma unroll
    for (int e = 0; e < 32; ++e) {
      int half = e >> 4, j = e & 15;
      int idx = aidx16(seg, lrow, half, j);
      float v = bf2f(Fhi[idx]) + bf2f(Flo[idx]);
      s += v; q += v * v;
    }
    red_s[lrow][seg] = s;
    red_q[lrow][seg] = q;
    __syncthreads();
    if (tid < 16) {
      float ss = 0.f, qq = 0.f;
#pragma unroll
      for (int g = 0; g < 16; ++g) { ss += red_s[tid][g]; qq += red_q[tid][g]; }
      float mu = ss * (1.f / 512.f);
      float var = qq * (1.f / 512.f) - mu * mu;
      mu_s[tid] = mu;
      rs_s[tid] = rsqrtf(var + 1e-5f);
    }
    __syncthreads();
    {
      float mu = mu_s[lrow], rs = rs_s[lrow];
#pragma unroll
      for (int e = 0; e < 32; ++e) {
        int half = e >> 4, j = e & 15;
        int idx = aidx16(seg, lrow, half, j);
        int col = seg * 32 + half * 8 + j + (j >= 8 ? 8 : 0);
        float v = bf2f(Fhi[idx]) + bf2f(Flo[idx]);
        v = (v - mu) * rs * gamma[col] + beta[col];
        outs[((size_t)(r0 + lrow) * 96 + t) * 512 + col] = v;
      }
    }
    __syncthreads();
  }
}

// ---------------------------------------------------------------------------
extern "C" void kernel_launch(void* const* d_in, const int* in_sizes, int n_in,
                              void* d_out, int out_size, void* d_ws,
                              size_t ws_size, hipStream_t stream) {
  (void)in_sizes; (void)n_in; (void)out_size; (void)ws_size;
  const float* x     = (const float*)d_in[0];
  const float* Wxh   = (const float*)d_in[1];
  const float* Whh   = (const float*)d_in[2];
  const float* gamma = (const float*)d_in[3];
  const float* beta  = (const float*)d_in[4];

  unsigned short* xh_hi = (unsigned short*)d_ws;       // 4 x 512KB fragments
  unsigned short* xh_lo = xh_hi + 262144;
  unsigned short* hh_hi = xh_lo + 262144;
  unsigned short* hh_lo = hh_hi + 262144;

  float* out  = (float*)d_out;
  float* rec  = out;                                   // [8,32,256,512]
  float* outs = out + (size_t)8 * 32 * 256 * 512;      // [8,32,96,512]

  prep_kernel<<<dim3(1024), 256, 0, stream>>>(Wxh, Whh, xh_hi, xh_lo,
                                              hh_hi, hh_lo);
  gemm_xw_kernel<<<dim3(512, 4), 256, 0, stream>>>(x, xh_hi, xh_lo, out);
  rec_kernel<<<dim3(16), 256, 0, stream>>>(hh_hi, hh_lo, rec);
  pred_kernel<<<dim3(16), 256, 0, stream>>>(hh_hi, hh_lo, gamma, beta,
                                            rec, outs);
}
